// HyperbolicDistanceHead_30270929502867
// MI455X (gfx1250) — compile-verified, matrix operand order checked by reading, not run on hardware
//
#include <hip/hip_runtime.h>

// ---------------------------------------------------------------------------
// HyperbolicDistanceHead for MI455X (gfx1250, wave32, WMMA).
//  d2 = xsq + ysq - 2*x.y^T  via bf16x3 split-precision WMMA GEMM.
//  - fp32->bf16(hi/lo) projection done ONCE in prep (if workspace allows),
//    removing all conversion VALU from the GEMM hot loop.
//  - double-buffered LDS; CDNA5 async global->LDS (ASYNCcnt) when available.
//  - fused arccosh epilogue, output streamed directly (512MB fp32).
// Weight bf16 hi/lo (131MB total) is L2-resident (192MB global L2).
// ---------------------------------------------------------------------------

typedef __attribute__((ext_vector_type(16))) __bf16 v16bf;
typedef __attribute__((ext_vector_type(8)))  __bf16 v8bf;
typedef __attribute__((ext_vector_type(4)))  __bf16 v4bf;
typedef __attribute__((ext_vector_type(8)))  float  v8f;
typedef __attribute__((ext_vector_type(4)))  int    v4i_t;

#define HD_EPS 1e-5f

#define TM 64    // block tile M
#define TN 128   // block tile N
#define TK 32    // K chunk == WMMA bf16 K
#define LDK 40   // padded LDS row (conflict-free b128 reads)

// ---- CDNA5 async global->LDS path, guarded so the file always compiles ----
#if defined(__HIP_DEVICE_COMPILE__) && __has_builtin(__builtin_amdgcn_global_load_async_to_lds_b128)
#define HD_ASYNC 1
#else
#define HD_ASYNC 0
#endif
#if defined(__HIP_DEVICE_COMPILE__) && __has_builtin(__builtin_amdgcn_s_wait_asynccnt)
#define HD_ASYNC_WAIT 1
#else
#define HD_ASYNC_WAIT 0
#endif

__device__ __forceinline__ void hd_cp16(const __bf16* g, __bf16* l) {
#if HD_ASYNC
  // C-style cast strips const; AS1/AS3 qualifications implicitly convert to
  // generic if the builtin's params are unqualified (AMDGPU AS lattice).
  v4i_t* gp = (v4i_t*)g;
  v4i_t* lp = (v4i_t*)l;
  __builtin_amdgcn_global_load_async_to_lds_b128(
      (__attribute__((address_space(1))) v4i_t*)gp,
      (__attribute__((address_space(3))) v4i_t*)lp, 0, 0);
#else
  *(v8bf*)l = *(const v8bf*)g;   // global_load_b128 + ds_store_b128
#endif
}

__device__ __forceinline__ void hd_wait_cp() {
#if HD_ASYNC
#if HD_ASYNC_WAIT
  __builtin_amdgcn_s_wait_asynccnt(0);
#else
  asm volatile("s_wait_asynccnt 0" ::: "memory");
#endif
#endif
}

// ---- per-row projection (+ optional fused bf16 hi/lo split) ---------------
__global__ void __launch_bounds__(256)
prep_rows(const float* __restrict__ src, int D,
          float* __restrict__ scale, float* __restrict__ sq,
          __bf16* __restrict__ hi, __bf16* __restrict__ lo) {
  const int row = blockIdx.x;
  const float* p = src + (size_t)row * D;
  float s = 0.f;
  for (int i = threadIdx.x; i < D; i += 256) {
    float v = p[i];
    s = fmaf(v, v, s);
  }
  for (int off = 16; off > 0; off >>= 1) s += __shfl_down(s, off, 32);
  __shared__ float part[8];
  __shared__ float sc;
  if ((threadIdx.x & 31) == 0) part[threadIdx.x >> 5] = s;
  __syncthreads();
  if (threadIdx.x == 0) {
    float t = 0.f;
#pragma unroll
    for (int i = 0; i < 8; ++i) t += part[i];
    float nrm = sqrtf(t);
    float c = fminf((1.0f - HD_EPS) / (nrm + 1e-10f), 1.0f);
    scale[row] = c;
    sq[row] = c * c * t;   // ||projected row||^2 in exact fp32
    sc = c;
  }
  __syncthreads();
  if (hi != nullptr) {     // fused projection + bf16x2 split, written once
    float c = sc;
    const float4* p4 = (const float4*)p;
    v4bf* h4 = (v4bf*)(hi + (size_t)row * D);
    v4bf* l4 = (v4bf*)(lo + (size_t)row * D);
    for (int t = threadIdx.x; t < (D >> 2); t += 256) {
      float4 v = p4[t];
      float f0 = v.x * c, f1 = v.y * c, f2 = v.z * c, f3 = v.w * c;
      __bf16 h0 = (__bf16)f0, h1 = (__bf16)f1, h2 = (__bf16)f2, h3 = (__bf16)f3;
      v4bf hv = {h0, h1, h2, h3};
      v4bf lv = {(__bf16)(f0 - (float)h0), (__bf16)(f1 - (float)h1),
                 (__bf16)(f2 - (float)h2), (__bf16)(f3 - (float)h3)};
      h4[t] = hv;
      l4[t] = lv;
    }
  }
}

// ---- fused GEMM + Poincare distance epilogue ------------------------------
template <bool PRE>
__global__ void __launch_bounds__(256)
hyp_gemm(const float* __restrict__ X, const float* __restrict__ W,
         const __bf16* __restrict__ Xh, const __bf16* __restrict__ Xl,
         const __bf16* __restrict__ Wh, const __bf16* __restrict__ Wl,
         const float* __restrict__ xscale, const float* __restrict__ xsq,
         const float* __restrict__ yscale, const float* __restrict__ ysq,
         float* __restrict__ out, int N, int V, int D) {
  __shared__ __bf16 Ah[2][TM][LDK];
  __shared__ __bf16 Al[2][TM][LDK];
  __shared__ __bf16 Bh[2][TN][LDK];
  __shared__ __bf16 Bl[2][TN][LDK];
  __shared__ float  sXsq[TM];
  __shared__ float  sYsq[TN];

  const int tid  = threadIdx.x;
  const int lane = tid & 31;
  const int wv   = tid >> 5;       // 0..7
  const int wm   = wv & 1;         // M offset wm*32
  const int wn   = wv >> 1;        // N offset wn*32
  const int rowBase = blockIdx.y * TM;
  const int colBase = blockIdx.x * TN;

  if (tid < TM) sXsq[tid] = xsq[min(rowBase + tid, N - 1)];
  if (tid < TN) sYsq[tid] = ysq[min(colBase + tid, V - 1)];

  const int hx = lane >> 4;        // K-half select per ISA A/B layout
  const int r  = lane & 15;

  v8f acc[2][2];
#pragma unroll
  for (int mi = 0; mi < 2; ++mi)
#pragma unroll
    for (int ni = 0; ni < 2; ++ni) {
      v8f z = {0.f, 0.f, 0.f, 0.f, 0.f, 0.f, 0.f, 0.f};
      acc[mi][ni] = z;
    }

  union Frag { v16bf v; struct { v8bf lo, hi; } h; };

  // -- stage precomputed bf16 tiles into buffer pb (async when available) --
  auto stagePre = [&](int pb, int k0) {
    {  // A: 64 rows x 4 b128-chunks = 256 -> 1 per thread per array
      int q = tid, rw = q >> 2, kq = q & 3;
      int gr = min(rowBase + rw, N - 1);
      size_t go = (size_t)gr * D + k0 + kq * 8;
      hd_cp16(Xh + go, &Ah[pb][rw][kq * 8]);
      hd_cp16(Xl + go, &Al[pb][rw][kq * 8]);
    }
#pragma unroll
    for (int t = 0; t < 2; ++t) {  // B: 128 rows x 4 chunks = 512 -> 2/thread
      int q = tid + t * 256, rw = q >> 2, kq = q & 3;
      int gr = min(colBase + rw, V - 1);
      size_t go = (size_t)gr * D + k0 + kq * 8;
      hd_cp16(Wh + go, &Bh[pb][rw][kq * 8]);
      hd_cp16(Wl + go, &Bl[pb][rw][kq * 8]);
    }
  };

  // -- fallback: fp32 load + on-the-fly scale/split (small-workspace path) --
  auto stageCvt = [&](int pb, int k0) {
#pragma unroll
    for (int t = 0; t < 2; ++t) {  // A: 512 float4
      int q = tid + t * 256, rw = q >> 3, kq = q & 7;
      int gr = min(rowBase + rw, N - 1);
      float4 vv = ((const float4*)(X + (size_t)gr * D + k0))[kq];
      float c = xscale[gr];
      float f0 = vv.x * c, f1 = vv.y * c, f2 = vv.z * c, f3 = vv.w * c;
      __bf16 h0 = (__bf16)f0, h1 = (__bf16)f1, h2 = (__bf16)f2, h3 = (__bf16)f3;
      v4bf hv = {h0, h1, h2, h3};
      v4bf lv = {(__bf16)(f0 - (float)h0), (__bf16)(f1 - (float)h1),
                 (__bf16)(f2 - (float)h2), (__bf16)(f3 - (float)h3)};
      *(v4bf*)&Ah[pb][rw][kq * 4] = hv;
      *(v4bf*)&Al[pb][rw][kq * 4] = lv;
    }
#pragma unroll
    for (int t = 0; t < 4; ++t) {  // B: 1024 float4
      int q = tid + t * 256, rw = q >> 3, kq = q & 7;
      int gr = min(colBase + rw, V - 1);
      float4 vv = ((const float4*)(W + (size_t)gr * D + k0))[kq];
      float c = yscale[gr];
      float f0 = vv.x * c, f1 = vv.y * c, f2 = vv.z * c, f3 = vv.w * c;
      __bf16 h0 = (__bf16)f0, h1 = (__bf16)f1, h2 = (__bf16)f2, h3 = (__bf16)f3;
      v4bf hv = {h0, h1, h2, h3};
      v4bf lv = {(__bf16)(f0 - (float)h0), (__bf16)(f1 - (float)h1),
                 (__bf16)(f2 - (float)h2), (__bf16)(f3 - (float)h3)};
      *(v4bf*)&Bh[pb][rw][kq * 4] = hv;
      *(v4bf*)&Bl[pb][rw][kq * 4] = lv;
    }
  };

  // -- fragments + 3 WMMAs per tile: hi*hi + lo*hi + hi*lo ------------------
  auto compute = [&](int pb) {
    Frag ah[2], al[2];
#pragma unroll
    for (int mi = 0; mi < 2; ++mi) {
      int rr = wm * 32 + mi * 16 + r;
      int ko = hx * 8;
      ah[mi].h.lo = *(const v8bf*)&Ah[pb][rr][ko];
      ah[mi].h.hi = *(const v8bf*)&Ah[pb][rr][ko + 16];
      al[mi].h.lo = *(const v8bf*)&Al[pb][rr][ko];
      al[mi].h.hi = *(const v8bf*)&Al[pb][rr][ko + 16];
    }
#pragma unroll
    for (int ni = 0; ni < 2; ++ni) {
      int rr = wn * 32 + ni * 16 + r;
      int ko = hx * 8;
      Frag bh, bl;
      bh.h.lo = *(const v8bf*)&Bh[pb][rr][ko];
      bh.h.hi = *(const v8bf*)&Bh[pb][rr][ko + 16];
      bl.h.lo = *(const v8bf*)&Bl[pb][rr][ko];
      bl.h.hi = *(const v8bf*)&Bl[pb][rr][ko + 16];
#pragma unroll
      for (int mi = 0; mi < 2; ++mi) {
        acc[mi][ni] = __builtin_amdgcn_wmma_f32_16x16x32_bf16(
            false, ah[mi].v, false, bh.v, (short)0, acc[mi][ni], false, false);
        acc[mi][ni] = __builtin_amdgcn_wmma_f32_16x16x32_bf16(
            false, al[mi].v, false, bh.v, (short)0, acc[mi][ni], false, false);
        acc[mi][ni] = __builtin_amdgcn_wmma_f32_16x16x32_bf16(
            false, ah[mi].v, false, bl.v, (short)0, acc[mi][ni], false, false);
      }
    }
  };

  const int nsteps = D / TK;
  if constexpr (PRE) {
    // software pipeline: stage(s+1) overlaps compute(s); one barrier/step
    stagePre(0, 0);
    hd_wait_cp();
    __syncthreads();
    for (int s = 0; s < nsteps; ++s) {
      int pb = s & 1;
      if (s + 1 < nsteps) stagePre(pb ^ 1, (s + 1) * TK);
      compute(pb);
      hd_wait_cp();
      __syncthreads();
    }
  } else {
    for (int s = 0; s < nsteps; ++s) {
      __syncthreads();
      stageCvt(0, s * TK);
      __syncthreads();
      compute(0);
    }
  }

  // ---- fused Poincare epilogue; C/D layout: lane->N, VGPR j -> M=j+8*hx
#pragma unroll
  for (int ni = 0; ni < 2; ++ni) {
    int nl = wn * 32 + ni * 16 + r;
    int ng = colBase + nl;
    float ysq_v = sYsq[nl];
    float oy = 1.0f - ysq_v;
#pragma unroll
    for (int mi = 0; mi < 2; ++mi) {
#pragma unroll
      for (int j = 0; j < 8; ++j) {
        int ml = wm * 32 + mi * 16 + hx * 8 + j;
        int mg = rowBase + ml;
        float xsq_v = sXsq[ml];
        float d2 = xsq_v + ysq_v - 2.0f * acc[mi][ni][j];
        float de = sqrtf(fmaxf(d2, 0.0f));
        float denom = (1.0f - xsq_v) * oy;
        float arg = 1.0f + 2.0f * de / (denom + HD_EPS);
        arg = fmaxf(arg, 1.0f + HD_EPS);
        float dist = __logf(arg + sqrtf(fmaxf(arg * arg - 1.0f, 0.0f)));
        if (mg < N && ng < V) out[(size_t)mg * V + ng] = -dist;
      }
    }
  }
}

// ---------------------------------------------------------------------------
extern "C" void kernel_launch(void* const* d_in, const int* in_sizes, int n_in,
                              void* d_out, int out_size, void* d_ws, size_t ws_size,
                              hipStream_t stream) {
  const float* X = (const float*)d_in[0];   // hidden_states (B*S, D) fp32
  const float* W = (const float*)d_in[1];   // weight        (V, D)   fp32
  float* out = (float*)d_out;               // logits (B*S, V) fp32

  const int D = 1024;
  const int N = in_sizes[0] / D;            // 4096
  const int V = in_sizes[1] / D;            // 32000

  // workspace layout: [xscale N][xsq N][yscale V][ysq V] | bf16 Xh Xl Wh Wl
  char* ws = (char*)d_ws;
  float* xscale = (float*)ws;
  float* xsq    = xscale + N;
  float* yscale = xsq + N;
  float* ysq    = yscale + V;
  size_t off = (((size_t)(2 * N + 2 * V)) * sizeof(float) + 255) & ~(size_t)255;

  const size_t szX = (size_t)N * D * sizeof(__bf16);
  const size_t szW = (size_t)V * D * sizeof(__bf16);
  __bf16* Xh = (__bf16*)(ws + off); off += szX;
  __bf16* Xl = (__bf16*)(ws + off); off += szX;
  __bf16* Wh = (__bf16*)(ws + off); off += szW;
  __bf16* Wl = (__bf16*)(ws + off); off += szW;

  const bool pre = (ws_size >= off);        // enough scratch for bf16 split?

  prep_rows<<<N, 256, 0, stream>>>(X, D, xscale, xsq,
                                   pre ? Xh : nullptr, pre ? Xl : nullptr);
  prep_rows<<<V, 256, 0, stream>>>(W, D, yscale, ysq,
                                   pre ? Wh : nullptr, pre ? Wl : nullptr);

  dim3 grid((V + TN - 1) / TN, (N + TM - 1) / TM);
  if (pre)
    hyp_gemm<true><<<grid, 256, 0, stream>>>(X, W, Xh, Xl, Wh, Wl,
                                             xscale, xsq, yscale, ysq, out, N, V, D);
  else
    hyp_gemm<false><<<grid, 256, 0, stream>>>(X, W, nullptr, nullptr, nullptr, nullptr,
                                              xscale, xsq, yscale, ysq, out, N, V, D);
}
